// PINN_47038481826198
// MI455X (gfx1250) — matrix operand order repaired and checked by reading, not compiled, not run on hardware
//
#include <hip/hip_runtime.h>
#include <math.h>

// ---------------------------------------------------------------------------
// Deep-set VMC log-psi evaluation for MI455X (gfx1250), wave32 + WMMA f16.
// Heavy math (psi pair-MLP, ~73 GFLOP) runs on v_wmma_f32_16x16x32_f16.
// Round 2: branch-free fast GELU/exp/log (libm erff/expf/log1pf were branchy
// and made the kernel VALU-bound ~10x over the matrix pipe).
// ---------------------------------------------------------------------------

typedef _Float16 half_t;
typedef __attribute__((ext_vector_type(16))) _Float16 v16h;
typedef __attribute__((ext_vector_type(8)))  _Float16 v8h;
typedef __attribute__((ext_vector_type(8)))  float    v8f;

#define NPART 64
#define NDIM  3
#define DLAT  5
#define NWALK 1024
#define NPAIR 2016   // 64*63/2

// per-wave LDS fence: make this wave's LDS stores visible before LDS reads
__device__ __forceinline__ void lds_fence() {
    asm volatile("s_wait_dscnt 0" ::: "memory");
}

// fast branch-free exp / log on v_exp_f32 / v_log_f32
__device__ __forceinline__ float fast_exp(float x) {
    return __builtin_amdgcn_exp2f(x * 1.4426950408889634f);
}
__device__ __forceinline__ float fast_log(float x) {
    return __builtin_amdgcn_logf(x) * 0.6931471805599453f;
}

// tanh-form GELU, branch-free, exp2 constants folded:
//   gelu(x) = x * (1 - 1/(1 + exp2(x*(2.3021144 + 0.10294426 x^2))))
// |error| ~1e-3 absolute, same order as f16 activation rounding.
__device__ __forceinline__ float gelu_f(float x) {
    float x2 = x * x;
    float t  = x * __builtin_fmaf(x2, 0.10294426f, 2.3021144f);
    float e  = __builtin_amdgcn_exp2f(t);
    float s  = __builtin_amdgcn_rcpf(1.0f + e);
    return __builtin_fmaf(-x, s, x);
}

// upper-triangle pair index -> (i,j), i<j, row-major.  S(i)=i*(127-i)/2
__device__ __forceinline__ void pair_ij(int p, int& i, int& j) {
    float disc = 16129.0f - 8.0f * (float)p;
    int ii = (int)((127.0f - sqrtf(disc)) * 0.5f);
    if (ii < 0) ii = 0;
    if (ii > 62) ii = 62;
    while ((ii + 1) * (127 - (ii + 1)) / 2 <= p) ++ii;
    while (ii * (127 - ii) / 2 > p) --ii;
    i = ii;
    j = p - ii * (127 - ii) / 2 + ii + 1;
}

// ---- WMMA fragment loads ---------------------------------------------------
// A (16x32 f16) gathered from an LDS row-major tile [16][KP] halves.
// Lane L: row m=L&15; lanes 0-15 hold K {0..7,16..23}, lanes 16-31 {8..15,24..31}.
__device__ __forceinline__ v16h load_a(const half_t* tile, int KP, int kt, int lane) {
    int m    = lane & 15;
    int base = (lane < 16) ? 0 : 8;
    const half_t* p = tile + m * KP + kt * 32 + base;
    v8h lo = *(const v8h*)(p);        // K = kt*32+base .. +7
    v8h hi = *(const v8h*)(p + 16);   // K = kt*32+base+16 .. +23
    return __builtin_shufflevector(lo, hi, 0,1,2,3,4,5,6,7,8,9,10,11,12,13,14,15);
}

// B (32x16 f16) pre-swizzled in global ws: frag is 512 halves, lane-contiguous.
__device__ __forceinline__ v16h load_b(const half_t* frag, int lane) {
    return *(const v16h*)(frag + lane * 16);
}

// One hidden layer: [16 x K] -> GELU -> [16 x 128], K = KT*32, NT = 8 n-tiles.
template<int KT, int KP>
__device__ __forceinline__ void mlp_hidden(const half_t* inT, half_t* outT,
                                           const half_t* Wf, const float* bias,
                                           int lane) {
    const int nlo = lane & 15;
    const int m0  = (lane < 16) ? 0 : 8;
#pragma unroll
    for (int nt = 0; nt < 8; ++nt) {
        float bv = bias[nt * 16 + nlo];
        v8f c = {bv, bv, bv, bv, bv, bv, bv, bv};
#pragma unroll
        for (int kt = 0; kt < KT; ++kt) {
            v16h a  = load_a(inT, KP, kt, lane);
            v16h bf = load_b(Wf + (size_t)(kt * 8 + nt) * 512, lane);
            c = __builtin_amdgcn_wmma_f32_16x16x32_f16(false, a, false, bf,
                                                       (short)0, c, false, false);
        }
#pragma unroll
        for (int r = 0; r < 8; ++r)
            outT[(m0 + r) * 128 + nt * 16 + nlo] = (half_t)gelu_f(c[r]);
    }
}

// Output layer: [16 x 128] -> [16 x nout<=16] (single n-tile), returns C frag.
template<int KT>
__device__ __forceinline__ v8f mlp_out16(const half_t* inT, const half_t* Wf,
                                         const float* bias, int nout, int lane) {
    const int nlo = lane & 15;
    float bv = (nlo < nout) ? bias[nlo] : 0.0f;
    v8f c = {bv, bv, bv, bv, bv, bv, bv, bv};
#pragma unroll
    for (int kt = 0; kt < KT; ++kt) {
        v16h a  = load_a(inT, 128, kt, lane);
        v16h bf = load_b(Wf + (size_t)kt * 512, lane);
        c = __builtin_amdgcn_wmma_f32_16x16x32_f16(false, a, false, bf,
                                                   (short)0, c, false, false);
    }
    return c;
}

// ---- Kernel 0: swizzle fp32 weights into per-lane-contiguous f16 B frags ---
__global__ void k_prep_frags(const float* __restrict__ W, half_t* __restrict__ dst,
                             int K, int Nout, int NT, int total) {
    int t = blockIdx.x * blockDim.x + threadIdx.x;
    if (t >= total) return;
    int f    = t >> 9;
    int lane = (t >> 4) & 31;
    int h    = t & 15;
    int kt = f / NT, nt = f % NT;
    int k = kt * 32 + ((lane < 16) ? 0 : 8) + ((h < 8) ? h : h + 8);
    int n = nt * 16 + (lane & 15);
    float v = (k < K && n < Nout) ? W[k * Nout + n] : 0.0f;
    dst[t] = (half_t)v;
}

// ---- Kernel 1: cheap pair pass: den[b,i], s1_mean[b], cusp[b] --------------
__global__ __launch_bounds__(256)
void k_pairs_light(const float* __restrict__ x, const int* __restrict__ spin,
                   float* __restrict__ den, float* __restrict__ s1m,
                   float* __restrict__ cusp) {
    const int b = blockIdx.x;
    const int tid = threadIdx.x;
    __shared__ float denL[NPART];
    __shared__ float accS[2];
    if (tid < NPART) denL[tid] = 0.0f;
    if (tid < 2) accS[tid] = 0.0f;
    __syncthreads();
    const float* xb = x + (size_t)b * NPART * NDIM;
    float s1p = 0.0f, cup = 0.0f;
    for (int p = tid; p < NPAIR; p += 256) {
        int i, j; pair_ij(p, i, j);
        float dx = xb[i*3+0] - xb[j*3+0];
        float dy = xb[i*3+1] - xb[j*3+1];
        float dz = xb[i*3+2] - xb[j*3+2];
        float r2 = dx*dx + dy*dy + dz*dz;
        float r  = sqrtf(r2 + 1.1920928955078125e-7f);
        float rt2 = (r2 + 0.04f) * 25.0f;           // (r2+eps^2)/eps^2, eps=0.2
        s1p += fast_log(1.0f + rt2);                // log1p, arg >= 1: safe
        float t  = r * 2.5f;                        // r/ATTN_RC
        float t2 = t * t;
        float w  = fast_exp(-(t2 * t2 * t2));       // exp(-(r/rc)^6)
        atomicAdd(&denL[i], w);
        float gamma = (spin[i] == spin[j]) ? 0.25f : 0.5f;
        cup += gamma * r * fast_exp(-r);            // CUSP_LEN = 1
    }
    for (int off = 16; off > 0; off >>= 1) {
        s1p += __shfl_down(s1p, off, 32);
        cup += __shfl_down(cup, off, 32);
    }
    if ((tid & 31) == 0) { atomicAdd(&accS[0], s1p); atomicAdd(&accS[1], cup); }
    __syncthreads();
    if (tid < NPART) den[b * NPART + tid] = denL[tid];
    if (tid == 0) s1m[b]  = accS[0] * (1.0f / NPAIR);
    if (tid == 1) cusp[b] = accS[1];
}

// ---- Kernel 2: phi per-particle MLP + phi_mean + r2_mean -------------------
__global__ __launch_bounds__(128)
void k_phi(const float* __restrict__ x,
           const half_t* wf0, const float* b0,
           const half_t* wf1, const float* b1,
           const half_t* wf2, const float* b2,
           float* __restrict__ phiM, float* __restrict__ r2m) {
    const int b    = blockIdx.x;
    const int lane = threadIdx.x & 31;
    const int wv   = threadIdx.x >> 5;
    __shared__ __align__(32) half_t bufA[4][16 * 128];
    __shared__ __align__(32) half_t bufB[4][16 * 128];
    __shared__ float acc[6];                 // 5 phi sums + r2 sum
    if (threadIdx.x < 6) acc[threadIdx.x] = 0.0f;
    __syncthreads();
    half_t* tA = bufA[wv];
    half_t* tB = bufB[wv];

    float r2part = 0.0f;
    if (lane < 16) {                         // one particle per lane
        int i = wv * 16 + lane;
        const float* xi = x + ((size_t)b * NPART + i) * 3;
        float x0 = xi[0], x1 = xi[1], x2 = xi[2];   // OMEGA = 1 -> x_scaled = x
        r2part = x0*x0 + x1*x1 + x2*x2;
        half_t* row = tB + lane * 32;
        row[0] = (half_t)x0; row[1] = (half_t)x1; row[2] = (half_t)x2;
        for (int c2 = 3; c2 < 32; ++c2) row[c2] = (half_t)0.0f;
    }
    lds_fence();
    mlp_hidden<1, 32 >(tB, tA, wf0, b0, lane);
    lds_fence();
    mlp_hidden<4, 128>(tA, tB, wf1, b1, lane);
    lds_fence();
    v8f c = mlp_out16<4>(tB, wf2, b2, DLAT, lane);

    float part = 0.0f;
#pragma unroll
    for (int r = 0; r < 8; ++r) part += c[r];         // sum over 8 rows
    part += __shfl_down(part, 16, 32);                // + other 8 rows
    int nlo = lane & 15;
    if (lane < 16 && nlo < DLAT) atomicAdd(&acc[nlo], part);
    for (int off = 16; off > 0; off >>= 1) r2part += __shfl_down(r2part, off, 32);
    if (lane == 0) atomicAdd(&acc[5], r2part);
    __syncthreads();
    if (threadIdx.x < DLAT) phiM[b * 8 + threadIdx.x] = acc[threadIdx.x] * (1.0f / NPART);
    if (threadIdx.x == 5)   r2m[b] = acc[5] * (1.0f / (NPART * NDIM));
}

// ---- Kernel 3: psi pair MLP (dominant ~73 GFLOP), gated+attention pooled ---
__global__ __launch_bounds__(128)
void k_psi(const float* __restrict__ x,
           const half_t* wf0, const float* b0,
           const half_t* wf1, const float* b1,
           const half_t* wf2, const float* b2,
           const float* __restrict__ den, float* __restrict__ psiM) {
    const int b    = blockIdx.x;
    const int lane = threadIdx.x & 31;
    const int wv   = threadIdx.x >> 5;
    __shared__ __align__(32) half_t bufA[4][16 * 128];
    __shared__ __align__(32) half_t bufB[4][16 * 128];
    __shared__ float coefS[4][16];
    __shared__ float acc[DLAT];
    if (threadIdx.x < DLAT) acc[threadIdx.x] = 0.0f;
    __syncthreads();
    half_t* tA = bufA[wv];
    half_t* tB = bufB[wv];
    float*  coef = coefS[wv];
    const float* xb   = x   + (size_t)b * NPART * 3;
    const float* denb = den + b * NPART;

    float part = 0.0f;                       // running column-partial (col = lane&15)
    for (int t = wv; t < NPAIR / 16; t += 4) {
        if (lane < 16) {
            int p = t * 16 + lane;
            int i, j; pair_ij(p, i, j);
            float dx = xb[i*3+0] - xb[j*3+0];
            float dy = xb[i*3+1] - xb[j*3+1];
            float dz = xb[i*3+2] - xb[j*3+2];
            float r2 = dx*dx + dy*dy + dz*dz;
            float r  = sqrtf(r2 + 1.1920928955078125e-7f);
            float rt2 = (r2 + 0.04f) * 25.0f;
            float s1 = fast_log(1.0f + rt2);
            float s2 = r2 / (r2 + 0.04f);
            float s3 = rt2 * fast_exp(-rt2);
            float rb0 = fast_exp(-0.25f * s1);
            float rb1 = fast_exp(-s1);
            float rb2 = fast_exp(-4.0f * s1);
            half_t* row = tB + lane * 32;
            row[0] = (half_t)s1;  row[1] = (half_t)s2;  row[2] = (half_t)s3;
            row[3] = (half_t)rb0; row[4] = (half_t)rb1; row[5] = (half_t)rb2;
            for (int c2 = 6; c2 < 32; ++c2) row[c2] = (half_t)0.0f;
            float gate = r2 / (r2 + 0.09f);            // GATE_R = 0.3
            float tt = r * 2.5f; float tt2 = tt * tt;
            float w  = fast_exp(-(tt2 * tt2 * tt2));
            coef[lane] = gate * w / (denb[i] + 1e-12f);
        }
        lds_fence();
        mlp_hidden<1, 32 >(tB, tA, wf0, b0, lane);
        lds_fence();
        mlp_hidden<4, 128>(tA, tB, wf1, b1, lane);
        lds_fence();
        v8f c = mlp_out16<4>(tB, wf2, b2, DLAT, lane);
        int m0 = (lane < 16) ? 0 : 8;
#pragma unroll
        for (int r = 0; r < 8; ++r) part += c[r] * coef[m0 + r];
        lds_fence();                         // before next tile rewrites coef/tB
    }
    part += __shfl_down(part, 16, 32);
    if (lane < DLAT) atomicAdd(&acc[lane], part);      // lanes 0..4, col = lane
    __syncthreads();
    if (threadIdx.x < DLAT) psiM[b * 8 + threadIdx.x] = acc[threadIdx.x] * (1.0f / NPART);
}

// ---- Kernel 4: rho MLP per walker + cusp -> output -------------------------
__global__ __launch_bounds__(128)
void k_rho(const float* __restrict__ phiM, const float* __restrict__ psiM,
           const float* __restrict__ r2m, const float* __restrict__ s1m,
           const float* __restrict__ cusp,
           const half_t* wf0, const float* b0,
           const half_t* wf1, const float* b1,
           const half_t* wf2, const float* b2,
           float* __restrict__ out) {
    const int lane = threadIdx.x & 31;
    const int wv   = threadIdx.x >> 5;
    const int tile = blockIdx.x * 4 + wv;    // 64 tiles of 16 walkers
    __shared__ __align__(32) half_t bufA[4][16 * 128];
    __shared__ __align__(32) half_t bufB[4][16 * 128];
    half_t* tA = bufA[wv];
    half_t* tB = bufB[wv];
    if (lane < 16) {
        int wk = tile * 16 + lane;
        half_t* row = tB + lane * 32;
        for (int l = 0; l < DLAT; ++l) row[l]        = (half_t)phiM[wk * 8 + l];
        for (int l = 0; l < DLAT; ++l) row[DLAT + l] = (half_t)psiM[wk * 8 + l];
        row[10] = (half_t)r2m[wk];
        row[11] = (half_t)s1m[wk];
        for (int c2 = 12; c2 < 32; ++c2) row[c2] = (half_t)0.0f;
    }
    lds_fence();
    mlp_hidden<1, 32 >(tB, tA, wf0, b0, lane);
    lds_fence();
    mlp_hidden<4, 128>(tA, tB, wf1, b1, lane);
    lds_fence();
    v8f c = mlp_out16<4>(tB, wf2, b2, 1, lane);
    int m0 = (lane < 16) ? 0 : 8;
    if ((lane & 15) == 0) {                  // column 0 only; lanes 0 and 16
#pragma unroll
        for (int r = 0; r < 8; ++r) {
            int wk = tile * 16 + m0 + r;
            out[wk] = c[r] + cusp[wk];
        }
    }
}

// ---------------------------------------------------------------------------
extern "C" void kernel_launch(void* const* d_in, const int* in_sizes, int n_in,
                              void* d_out, int out_size, void* d_ws, size_t ws_size,
                              hipStream_t stream) {
    (void)in_sizes; (void)n_in; (void)out_size; (void)ws_size;
    const float* x      = (const float*)d_in[0];
    const int*   spin   = (const int*)d_in[1];
    const float* phi_w0 = (const float*)d_in[2];  const float* phi_b0 = (const float*)d_in[3];
    const float* phi_w1 = (const float*)d_in[4];  const float* phi_b1 = (const float*)d_in[5];
    const float* phi_w2 = (const float*)d_in[6];  const float* phi_b2 = (const float*)d_in[7];
    const float* psi_w0 = (const float*)d_in[8];  const float* psi_b0 = (const float*)d_in[9];
    const float* psi_w1 = (const float*)d_in[10]; const float* psi_b1 = (const float*)d_in[11];
    const float* psi_w2 = (const float*)d_in[12]; const float* psi_b2 = (const float*)d_in[13];
    const float* rho_w0 = (const float*)d_in[14]; const float* rho_b0 = (const float*)d_in[15];
    const float* rho_w1 = (const float*)d_in[16]; const float* rho_b1 = (const float*)d_in[17];
    const float* rho_w2 = (const float*)d_in[18]; const float* rho_b2 = (const float*)d_in[19];

    char* ws = (char*)d_ws;
    // f16 weight fragments: 3 MLPs x (8 + 32 + 4) frags x 512 halves
    half_t* fPhi0 = (half_t*)(ws);
    half_t* fPhi1 = fPhi0 + 8  * 512;
    half_t* fPhi2 = fPhi1 + 32 * 512;
    half_t* fPsi0 = fPhi2 + 4  * 512;
    half_t* fPsi1 = fPsi0 + 8  * 512;
    half_t* fPsi2 = fPsi1 + 32 * 512;
    half_t* fRho0 = fPsi2 + 4  * 512;
    half_t* fRho1 = fRho0 + 8  * 512;
    half_t* fRho2 = fRho1 + 32 * 512;
    float* den  = (float*)(ws + 135168);   // [B][64]
    float* s1m  = (float*)(ws + 397312);   // [B]
    float* cusp = (float*)(ws + 401408);   // [B]
    float* phiM = (float*)(ws + 405504);   // [B][8]
    float* psiM = (float*)(ws + 438272);   // [B][8]
    float* r2m  = (float*)(ws + 471040);   // [B]

    auto prep = [&](const float* W, half_t* dst, int K, int Nout, int KT, int NT) {
        int total = KT * NT * 512;
        k_prep_frags<<<(total + 255) / 256, 256, 0, stream>>>(W, dst, K, Nout, NT, total);
    };
    prep(phi_w0, fPhi0, 3,   128, 1, 8);
    prep(phi_w1, fPhi1, 128, 128, 4, 8);
    prep(phi_w2, fPhi2, 128, 5,   4, 1);
    prep(psi_w0, fPsi0, 6,   128, 1, 8);
    prep(psi_w1, fPsi1, 128, 128, 4, 8);
    prep(psi_w2, fPsi2, 128, 5,   4, 1);
    prep(rho_w0, fRho0, 12,  128, 1, 8);
    prep(rho_w1, fRho1, 128, 128, 4, 8);
    prep(rho_w2, fRho2, 128, 1,   4, 1);

    k_pairs_light<<<NWALK, 256, 0, stream>>>(x, spin, den, s1m, cusp);
    k_phi<<<NWALK, 128, 0, stream>>>(x, fPhi0, phi_b0, fPhi1, phi_b1, fPhi2, phi_b2,
                                     phiM, r2m);
    k_psi<<<NWALK, 128, 0, stream>>>(x, fPsi0, psi_b0, fPsi1, psi_b1, fPsi2, psi_b2,
                                     den, psiM);
    k_rho<<<16, 128, 0, stream>>>(phiM, psiM, r2m, s1m, cusp,
                                  fRho0, rho_b0, fRho1, rho_b1, fRho2, rho_b2,
                                  (float*)d_out);
}
